// DRNLFilterbank_28613072126397
// MI455X (gfx1250) — matrix operand order repaired and checked by reading, not compiled, not run on hardware
//
// DRNL filterbank for MI455X (gfx1250): WMMA bf16 hi/lo Toeplitz-FIR + scan-parallel IIR.
#include <hip/hip_runtime.h>

typedef __bf16 bf16_t;
typedef __attribute__((ext_vector_type(16))) __bf16 v16bf;
typedef __attribute__((ext_vector_type(8)))  float  v8f;

#define DEVFN __device__ __forceinline__

constexpr int B_   = 8;
constexpr int T_   = 20000;
constexpr int C_   = 50;
constexpr int KF   = 1000;     // gammatone FIR taps
constexpr int KME  = 512;      // middle-ear FIR taps
constexpr int LP   = 21312;    // padded per-sequence bf16 signal length
constexpr int SPAD = 1023;     // left zero pad (keeps per-lane reads dword aligned)
constexpr int NTILE = 79;      // ceil(20000/256) time tiles (256 outputs per WMMA chain)
constexpr int NKCH  = 32;      // 32 k-chunks of 32 -> q in [0,1024) covers K+15=1015
constexpr int SZ_AF = C_ * NKCH * 512;   // elements per A-fragment set (16x32 tile = 512)
constexpr int LCHUNK = 160;    // IIR scan chunk length
constexpr int NCHUNK = 125;    // 125*160 = 20000
constexpr float IN_GAIN = 0.500034535f;  // 10^((93.98-100)/20)

// ---------------------------------------------------------------- WMMA helpers
DEVFN v8f wmma_bf(v16bf a, v16bf b, v8f c) {
  // v_wmma_f32_16x16x32_bf16  (neg_a, A, neg_b, B, c_mod, C, reuse_a, reuse_b)
  return __builtin_amdgcn_wmma_f32_16x16x32_bf16(false, a, false, b, (short)0, c, false, false);
}
DEVFN v8f vzero() {
  v8f z;
  #pragma unroll
  for (int r = 0; r < 8; ++r) z[r] = 0.f;
  return z;
}
// A fragments precomputed in per-lane-contiguous order: 32B aligned per lane.
DEVFN v16bf load_af(const bf16_t* af, int c, int chunk, int lane) {
  return *reinterpret_cast<const v16bf*>(af + ((((c * NKCH) + chunk) * 32 + lane) << 4));
}
// B fragment: lane (n=lane&15, h=lane>>4) holds K = e+16h  ->  16 contiguous bf16
// at element i0 = SPAD + T0 + 32*chunk - 999 + 16*(n+h)  (always even -> dword aligned).
DEVFN v16bf load_bf(const bf16_t* sig, int i0) {
  union { unsigned int u[8]; v16bf v; } t;
  const unsigned int* p = reinterpret_cast<const unsigned int*>(sig + i0);
  #pragma unroll
  for (int d = 0; d < 8; ++d) t.u[d] = p[d];
  return t.v;
}

// ---------------------------------------------------------------- K1: middle ear FIR (f32 VALU, 0.3% of FLOPs)
__global__ void k_me_fir(const float* __restrict__ x, const float* __restrict__ me,
                         float* __restrict__ xme) {
  __shared__ float sx[768];
  __shared__ float sm[KME];
  const int b  = blockIdx.y;
  const int t0 = blockIdx.x * 256;
  for (int j = threadIdx.x; j < 768; j += 256) {
    int idx = t0 - 512 + j;
    sx[j] = (idx >= 0 && idx < T_) ? IN_GAIN * x[b * T_ + idx] : 0.f;
  }
  for (int j = threadIdx.x; j < KME; j += 256) sm[j] = me[j];
  __syncthreads();
  float acc = 0.f;
  #pragma unroll 8
  for (int k = 0; k < KME; ++k) acc = fmaf(sm[k], sx[512 + (int)threadIdx.x - k], acc);
  int t = t0 + threadIdx.x;
  if (t < T_) xme[b * T_ + t] = acc;
}

// ---------------------------------------------------------------- K2: pack xme -> padded bf16 hi/lo
__global__ void k_pack_x(const float* __restrict__ xme,
                         bf16_t* __restrict__ xh, bf16_t* __restrict__ xl) {
  int idx = blockIdx.x * 256 + threadIdx.x;
  if (idx >= B_ * LP) return;
  int b = idx / LP, i = idx - b * LP, t = i - SPAD;
  float v = (t >= 0 && t < T_) ? xme[b * T_ + t] : 0.f;
  bf16_t h = (bf16_t)v;
  xh[idx] = h;
  xl[idx] = (bf16_t)(v - (float)h);
}

// ---------------------------------------------------------------- K3: filters -> Toeplitz A fragments (hi/lo)
// A[m,q] = f[999+m-q]; A layout matches B: lane<16: m=lane, K in {0..15};
// lane>=16: m=lane-16, K+16.  Sets: 0/1 lin(hi/lo, gain folded) 2/3 nlin-before 4/5 nlin-after.
__global__ void k_prep_filters(const float* __restrict__ linf, const float* __restrict__ nbf,
                               const float* __restrict__ naf, const float* __restrict__ gain,
                               bf16_t* __restrict__ af) {
  int tid = blockIdx.x * 256 + threadIdx.x;
  if (tid >= C_ * NKCH * 32) return;
  const int lane  = tid & 31;
  const int chunk = (tid >> 5) & 31;
  const int c     = tid >> 10;
  const int m     = lane & 15;
  const int base8 = (lane >> 4) * 8;
  const float g   = gain[c];
  const size_t o  = (size_t)tid * 16;     // ((c*32+chunk)*32+lane)*16
  #pragma unroll
  for (int e = 0; e < 16; ++e) {
    int v    = e >> 1;
    int kloc = e + 2 * base8; (void)v;
    int q    = chunk * 32 + kloc;
    int fi   = (KF - 1) + m - q;
    float fl = 0.f, fb = 0.f, fa = 0.f;
    if (fi >= 0 && fi < KF) {
      fl = g * linf[c * KF + fi];
      fb = nbf[c * KF + fi];
      fa = naf[c * KF + fi];
    }
    bf16_t h;
    h = (bf16_t)fl; af[0 * SZ_AF + o + e] = h; af[1 * SZ_AF + o + e] = (bf16_t)(fl - (float)h);
    h = (bf16_t)fb; af[2 * SZ_AF + o + e] = h; af[3 * SZ_AF + o + e] = (bf16_t)(fb - (float)h);
    h = (bf16_t)fa; af[4 * SZ_AF + o + e] = h; af[5 * SZ_AF + o + e] = (bf16_t)(fa - (float)h);
  }
}

// ---------------------------------------------------------------- K4: fused lin + nlin-before FIR (shared B)
// One wave per (channel, time-tile); A reused across all 8 batches; 48 WMMA / k-chunk.
__global__ void __launch_bounds__(32) k_conv_shared(const bf16_t* __restrict__ af,
                                                    const bf16_t* __restrict__ xh,
                                                    const bf16_t* __restrict__ xl,
                                                    float* __restrict__ linpre,
                                                    float* __restrict__ nmid) {
  const int c    = blockIdx.x % C_;
  const int tile = blockIdx.x / C_;
  const int lane = threadIdx.x;
  const int nph  = (lane & 15) + (lane >> 4);
  v8f accL[B_], accN[B_];
  #pragma unroll
  for (int b = 0; b < B_; ++b) { accL[b] = vzero(); accN[b] = vzero(); }
  const int ib0 = SPAD + tile * 256 - (KF - 1) + 16 * nph;
  for (int chunk = 0; chunk < NKCH; ++chunk) {
    v16bf aLH = load_af(af + 0 * SZ_AF, c, chunk, lane);
    v16bf aLL = load_af(af + 1 * SZ_AF, c, chunk, lane);
    v16bf aNH = load_af(af + 2 * SZ_AF, c, chunk, lane);
    v16bf aNL = load_af(af + 3 * SZ_AF, c, chunk, lane);
    if (chunk + 1 < NKCH) __builtin_prefetch(af + 0 * SZ_AF + (((c * NKCH) + chunk + 1) * 32 + lane) * 16, 0, 0);
    const int i0 = ib0 + chunk * 32;
    #pragma unroll
    for (int b = 0; b < B_; ++b) {
      v16bf bH = load_bf(xh + b * LP, i0);
      v16bf bL = load_bf(xl + b * LP, i0);
      accL[b] = wmma_bf(aLH, bH, accL[b]);
      accL[b] = wmma_bf(aLH, bL, accL[b]);
      accL[b] = wmma_bf(aLL, bH, accL[b]);
      accN[b] = wmma_bf(aNH, bH, accN[b]);
      accN[b] = wmma_bf(aNH, bL, accN[b]);
      accN[b] = wmma_bf(aNL, bH, accN[b]);
    }
  }
  const int tb = tile * 256 + 16 * (lane & 15) + 8 * (lane >> 4);  // t = T0+16n+m, m=r+8h
  #pragma unroll
  for (int b = 0; b < B_; ++b) {
    const size_t ob = (size_t)(b * C_ + c) * T_;
    #pragma unroll
    for (int r = 0; r < 8; ++r) {
      int t = tb + r;
      if (t < T_) { linpre[ob + t] = accL[b][r]; nmid[ob + t] = accN[b][r]; }
    }
  }
}

// ---------------------------------------------------------------- K5: compressive nonlinearity + pack per-seq bf16 hi/lo
__global__ void k_nonlin_pack(const float* __restrict__ nmid,
                              const float* __restrict__ na, const float* __restrict__ nb,
                              const float* __restrict__ nc,
                              bf16_t* __restrict__ uh, bf16_t* __restrict__ ul) {
  long idx = (long)blockIdx.x * 256 + threadIdx.x;
  if (idx >= (long)B_ * C_ * LP) return;
  int s = (int)(idx / LP);
  int i = (int)(idx - (long)s * LP);
  int t = i - SPAD;
  float v = 0.f;
  if (t >= 0 && t < T_) {
    int c = s % C_;
    float y  = nmid[(size_t)s * T_ + t];
    float ay = fabsf(y);
    float m  = fminf(na[c] * ay, nb[c] * powf(fmaxf(ay, 1e-12f), nc[c]));
    v = (y < 0.f) ? -m : m;
  }
  bf16_t h = (bf16_t)v;
  uh[idx] = h;
  ul[idx] = (bf16_t)(v - (float)h);
}

// ---------------------------------------------------------------- K6: grouped nlin-after FIR (per-seq B)
__global__ void __launch_bounds__(32) k_conv_grouped(const bf16_t* __restrict__ af,
                                                     const bf16_t* __restrict__ uh,
                                                     const bf16_t* __restrict__ ul,
                                                     float* __restrict__ nconv) {
  const int c    = blockIdx.x % C_;
  const int tile = blockIdx.x / C_;
  const int lane = threadIdx.x;
  const int nph  = (lane & 15) + (lane >> 4);
  v8f acc[B_];
  #pragma unroll
  for (int b = 0; b < B_; ++b) acc[b] = vzero();
  const int ib0 = SPAD + tile * 256 - (KF - 1) + 16 * nph;
  for (int chunk = 0; chunk < NKCH; ++chunk) {
    v16bf aH = load_af(af + 4 * SZ_AF, c, chunk, lane);
    v16bf aL = load_af(af + 5 * SZ_AF, c, chunk, lane);
    const int i0 = ib0 + chunk * 32;
    #pragma unroll
    for (int b = 0; b < B_; ++b) {
      const size_t sb = (size_t)(b * C_ + c) * LP;
      v16bf bH = load_bf(uh + sb, i0);
      v16bf bL = load_bf(ul + sb, i0);
      acc[b] = wmma_bf(aH, bH, acc[b]);
      acc[b] = wmma_bf(aH, bL, acc[b]);
      acc[b] = wmma_bf(aL, bH, acc[b]);
    }
  }
  const int tb = tile * 256 + 16 * (lane & 15) + 8 * (lane >> 4);
  #pragma unroll
  for (int b = 0; b < B_; ++b) {
    const size_t ob = (size_t)(b * C_ + c) * T_;
    #pragma unroll
    for (int r = 0; r < 8; ++r) {
      int t = tb + r;
      if (t < T_) nconv[ob + t] = acc[b][r];
    }
  }
}

// ---------------------------------------------------------------- IIR cascade: scan-parallel (DF2T, constant 2x2 blocks)
template <int NS>
DEVFN float casc_step(float x, float* z1, float* z2,
                      float b0, float b1, float b2, float a1, float a2) {
  #pragma unroll
  for (int s = 0; s < NS; ++s) {
    float y   = fmaf(b0, x, z1[s]);
    float nz1 = fmaf(b1, x, fmaf(-a1, y, z2[s]));
    float nz2 = fmaf(b2, x, -a2 * y);
    z1[s] = nz1; z2[s] = nz2; x = y;
  }
  return x;
}

// pass1: zero-state response per chunk + final states
template <int NS, bool ADD>
__global__ void k_iir_pass1(const float* __restrict__ in, float* __restrict__ out,
                            float* __restrict__ fst,
                            const float* __restrict__ bco, const float* __restrict__ aco) {
  int tid = blockIdx.x * 256 + threadIdx.x;
  if (tid >= B_ * C_ * NCHUNK) return;
  int s = tid / NCHUNK, ch = tid - s * NCHUNK, c = s % C_;
  float b0 = bco[c * 3 + 0], b1 = bco[c * 3 + 1], b2 = bco[c * 3 + 2];
  float a1 = aco[c * 3 + 1], a2 = aco[c * 3 + 2];
  float z1[NS], z2[NS];
  #pragma unroll
  for (int k = 0; k < NS; ++k) { z1[k] = 0.f; z2[k] = 0.f; }
  size_t base = (size_t)s * T_ + (size_t)ch * LCHUNK;
  for (int j = 0; j < LCHUNK; ++j) {
    float y = casc_step<NS>(in[base + j], z1, z2, b0, b1, b2, a1, a2);
    if (ADD) out[base + j] += y; else out[base + j] = y;
  }
  float* f = fst + (size_t)tid * 2 * NS;
  #pragma unroll
  for (int k = 0; k < NS; ++k) { f[2 * k] = z1[k]; f[2 * k + 1] = z2[k]; }
}

template <int D>
DEVFN void mm(float (*A)[D], float (*Bm)[D], float (*Cm)[D]) {
  for (int i = 0; i < D; ++i)
    for (int j = 0; j < D; ++j) {
      float acc = 0.f;
      for (int k = 0; k < D; ++k) acc = fmaf(A[i][k], Bm[k][j], acc);
      Cm[i][j] = acc;
    }
}

// pass2: sequential over 125 chunks only — s_{c+1} = Phi*s_c + f_c, Phi = A_cascade^160
template <int NS>
__global__ void k_iir_fix(const float* __restrict__ fst, float* __restrict__ sst,
                          const float* __restrict__ bco, const float* __restrict__ aco) {
  int tid = blockIdx.x * 256 + threadIdx.x;
  if (tid >= B_ * C_) return;
  int c = tid % C_;
  float b0 = bco[c * 3 + 0], b1 = bco[c * 3 + 1], b2 = bco[c * 3 + 2];
  float a1 = aco[c * 3 + 1], a2 = aco[c * 3 + 2];
  constexpr int D = 2 * NS;
  float M[D][D];
  for (int j = 0; j < D; ++j) {        // columns = one homogeneous step from e_j
    float z1[NS], z2[NS];
    #pragma unroll
    for (int k = 0; k < NS; ++k) { z1[k] = 0.f; z2[k] = 0.f; }
    if (j & 1) z2[j >> 1] = 1.f; else z1[j >> 1] = 1.f;
    casc_step<NS>(0.f, z1, z2, b0, b1, b2, a1, a2);
    for (int k = 0; k < NS; ++k) { M[2 * k][j] = z1[k]; M[2 * k + 1][j] = z2[k]; }
  }
  float R[D][D], Bp[D][D], Tt[D][D];
  for (int i = 0; i < D; ++i)
    for (int j = 0; j < D; ++j) { R[i][j] = (i == j) ? 1.f : 0.f; Bp[i][j] = M[i][j]; }
  int e = LCHUNK;                      // Phi = M^160 by squaring
  while (e) {
    if (e & 1) { mm<D>(R, Bp, Tt); for (int i = 0; i < D; ++i) for (int j = 0; j < D; ++j) R[i][j] = Tt[i][j]; }
    e >>= 1;
    if (e) { mm<D>(Bp, Bp, Tt); for (int i = 0; i < D; ++i) for (int j = 0; j < D; ++j) Bp[i][j] = Tt[i][j]; }
  }
  float sv[D], nv[D];
  for (int i = 0; i < D; ++i) sv[i] = 0.f;
  for (int ch = 0; ch < NCHUNK; ++ch) {
    float* sp = sst + ((size_t)tid * NCHUNK + ch) * D;
    const float* fp = fst + ((size_t)tid * NCHUNK + ch) * D;
    for (int i = 0; i < D; ++i) {
      sp[i] = sv[i];
      float acc = fp[i];
      for (int k = 0; k < D; ++k) acc = fmaf(R[i][k], sv[k], acc);
      nv[i] = acc;
    }
    for (int i = 0; i < D; ++i) sv[i] = nv[i];
  }
}

// pass3: add homogeneous response of true chunk-initial states
template <int NS>
__global__ void k_iir_corr(const float* __restrict__ sst, float* __restrict__ out,
                           const float* __restrict__ bco, const float* __restrict__ aco) {
  int tid = blockIdx.x * 256 + threadIdx.x;
  if (tid >= B_ * C_ * NCHUNK) return;
  int s = tid / NCHUNK, ch = tid - s * NCHUNK, c = s % C_;
  float b0 = bco[c * 3 + 0], b1 = bco[c * 3 + 1], b2 = bco[c * 3 + 2];
  float a1 = aco[c * 3 + 1], a2 = aco[c * 3 + 2];
  float z1[NS], z2[NS];
  const float* sp = sst + (size_t)tid * 2 * NS;
  #pragma unroll
  for (int k = 0; k < NS; ++k) { z1[k] = sp[2 * k]; z2[k] = sp[2 * k + 1]; }
  size_t base = (size_t)s * T_ + (size_t)ch * LCHUNK;
  for (int j = 0; j < LCHUNK; ++j) {
    float y = casc_step<NS>(0.f, z1, z2, b0, b1, b2, a1, a2);
    out[base + j] += y;
  }
}

// ---------------------------------------------------------------- host
extern "C" void kernel_launch(void* const* d_in, const int* in_sizes, int n_in,
                              void* d_out, int out_size, void* d_ws, size_t ws_size,
                              hipStream_t stream) {
  const float* x      = (const float*)d_in[0];
  const float* me     = (const float*)d_in[1];
  const float* linfir = (const float*)d_in[2];
  const float* nbfir  = (const float*)d_in[3];
  const float* nafir  = (const float*)d_in[4];
  const float* lb     = (const float*)d_in[5];
  const float* la     = (const float*)d_in[6];
  const float* nqb    = (const float*)d_in[7];
  const float* nqa    = (const float*)d_in[8];
  const float* gain   = (const float*)d_in[9];
  const float* nla    = (const float*)d_in[10];
  const float* nlb    = (const float*)d_in[11];
  const float* nlc    = (const float*)d_in[12];
  float* out = (float*)d_out;
  (void)in_sizes; (void)n_in; (void)out_size; (void)ws_size;

  char* w = (char*)d_ws;
  size_t cur = 0;
  auto alloc = [&](size_t bytes) -> char* {
    char* p = w + cur;
    cur += bytes;
    cur = (cur + 255) & ~(size_t)255;
    return p;
  };
  float*  xme    = (float*) alloc((size_t)B_ * T_ * 4);
  bf16_t* xpH    = (bf16_t*)alloc((size_t)B_ * LP * 2);
  bf16_t* xpL    = (bf16_t*)alloc((size_t)B_ * LP * 2);
  bf16_t* af     = (bf16_t*)alloc((size_t)6 * SZ_AF * 2);
  float*  linpre = (float*) alloc((size_t)B_ * C_ * T_ * 4);
  float*  nmid   = (float*) alloc((size_t)B_ * C_ * T_ * 4);
  bf16_t* upH    = (bf16_t*)alloc((size_t)B_ * C_ * LP * 2);
  bf16_t* upL    = (bf16_t*)alloc((size_t)B_ * C_ * LP * 2);
  float*  fst    = (float*) alloc((size_t)B_ * C_ * NCHUNK * 8 * 4);
  float*  sst    = (float*) alloc((size_t)B_ * C_ * NCHUNK * 8 * 4);
  float*  nconv  = nmid;  // nmid is dead after K5; reuse for the nlin-after conv output

  const int nseqchunks = B_ * C_ * NCHUNK;  // 50000

  k_me_fir<<<dim3(NTILE, B_), 256, 0, stream>>>(x, me, xme);
  k_pack_x<<<(B_ * LP + 255) / 256, 256, 0, stream>>>(xme, xpH, xpL);
  k_prep_filters<<<(C_ * NKCH * 32 + 255) / 256, 256, 0, stream>>>(linfir, nbfir, nafir, gain, af);
  k_conv_shared<<<C_ * NTILE, 32, 0, stream>>>(af, xpH, xpL, linpre, nmid);
  k_nonlin_pack<<<(int)(((size_t)B_ * C_ * LP + 255) / 256), 256, 0, stream>>>(nmid, nla, nlb, nlc, upH, upL);
  k_conv_grouped<<<C_ * NTILE, 32, 0, stream>>>(af, upH, upL, nconv);

  // linear path: 4-stage cascade -> writes d_out
  k_iir_pass1<4, false><<<(nseqchunks + 255) / 256, 256, 0, stream>>>(linpre, out, fst, lb, la);
  k_iir_fix<4><<<2, 256, 0, stream>>>(fst, sst, lb, la);
  k_iir_corr<4><<<(nseqchunks + 255) / 256, 256, 0, stream>>>(sst, out, lb, la);
  // nonlinear path: 3-stage cascade -> accumulates into d_out
  k_iir_pass1<3, true><<<(nseqchunks + 255) / 256, 256, 0, stream>>>(nconv, out, fst, nqb, nqa);
  k_iir_fix<3><<<2, 256, 0, stream>>>(fst, sst, nqb, nqa);
  k_iir_corr<3><<<(nseqchunks + 255) / 256, 256, 0, stream>>>(sst, out, nqb, nqa);
}
